// SCAC_DS_Attn_30597347017342
// MI455X (gfx1250) — matrix operand order repaired and checked by reading, not compiled, not run on hardware
//
#include <hip/hip_runtime.h>
#include <hip/hip_fp16.h>

typedef _Float16 v16h __attribute__((ext_vector_type(16)));
typedef _Float16 v8h  __attribute__((ext_vector_type(8)));
typedef float    v8f  __attribute__((ext_vector_type(8)));

#define NB 8
#define NC 48
#define NO 96
#define CO 4608          // NC*NO
#define NL 1024
#define CLEN_ 8
#define NCOND 12
#define NTILE 288        // CO/16
#define SSTR 4616        // padded logit row stride in halfs (4608 + 8)
#define YSTR 97          // padded output-tile row stride in floats

// ---------- K1: kD = Wk_D @ kf, vD = Wv_D @ kf  (9x9 mat-vec) ----------
__global__ void k_prep_small(const float* __restrict__ WkD,
                             const float* __restrict__ WvD,
                             const float* __restrict__ kernelD,
                             float* __restrict__ kD, float* __restrict__ vD) {
  int t = threadIdx.x;
  if (t < 9) {
    float sk = 0.f, sv = 0.f;
    for (int j = 0; j < 9; ++j) {
      float kf = kernelD[j];
      sk += WkD[t * 9 + j] * kf;
      sv += WvD[t * 9 + j] * kf;
    }
    kD[t] = sk;
    vD[t] = sv;
  }
}

// ---------- K2: kP = Wk_P @ kernel_P.flat, vP = Wv_P @ kernel_P.flat ----------
// One wave per output row; 2*4608 = 9216 row tasks; HBM-bound (170 MB of weights).
__global__ void k_matvec_big(const float* __restrict__ WkP,
                             const float* __restrict__ WvP,
                             const float* __restrict__ kernelP,
                             float* __restrict__ kP, float* __restrict__ vP) {
  int gw   = (blockIdx.x * blockDim.x + threadIdx.x) >> 5;
  int lane = threadIdx.x & 31;
  const float* W;
  float* dst;
  int row;
  if (gw < CO) { W = WkP; dst = kP; row = gw; }
  else         { W = WvP; dst = vP; row = gw - CO; }
  const float* wr = W + (size_t)row * CO;
  float s = 0.f;
  for (int i = lane; i < CO; i += 32) s += wr[i] * kernelP[i];
  #pragma unroll
  for (int m = 16; m >= 1; m >>= 1) s += __shfl_xor(s, m, 32);
  if (lane == 0) dst[row] = s;
}

// ---------- K3: pack A-operand tiles: A[co,k] = kP[co] * Wq_P[co,k] (f16, K padded 12->32)
// Layout per v_wmma_f32_16x16x32_f16 A operand (16-bit A 16x32), contiguous K order:
//   lanes 0-15  (M=lane):    halfs[0..15] = K 0..15 (only 0..11 valid)
//   lanes 16-31 (M=lane-16): halfs[0..15] = K16..31 (zero)
__global__ void k_pack_A(const float* __restrict__ WqP,
                         const float* __restrict__ kP,
                         _Float16* __restrict__ Apack) {
  int T    = blockIdx.x;       // 0..287
  int lane = threadIdx.x;      // 0..31
  int m    = lane & 15;
  int row  = T * 16 + m;
  float scale = kP[row];
  const float* wr = WqP + (size_t)row * NCOND;
  _Float16 h[16];
  #pragma unroll
  for (int i = 0; i < 16; ++i) h[i] = (_Float16)0.f;
  if (lane < 16) {
    #pragma unroll
    for (int i = 0; i < NCOND; ++i) h[i] = (_Float16)(wr[i] * scale);    // K=0..11
  }
  _Float16* o = Apack + (((size_t)T * 32 + lane) << 4);
  #pragma unroll
  for (int i = 0; i < 16; ++i) o[i] = h[i];
}

// ---------- K4: depthwise adaptive conv -> xs[B, C, L] ----------
__global__ void k_depthwise(const float* __restrict__ x,
                            const float* __restrict__ kernelD,
                            const float* __restrict__ WqD,
                            const float* __restrict__ kD,
                            const float* __restrict__ vD,
                            const float* __restrict__ a,
                            const float* __restrict__ cpar,
                            float* __restrict__ xs) {
  int g = blockIdx.x * blockDim.x + threadIdx.x;   // 0..8191
  int b = g >> 10, l = g & 1023;

  float cond[NCOND];
  cond[0] = (float)(l >> 5) * (1.f / 31.f);   // matches reference's meshgrid quirk
  cond[1] = 1.f - cond[0];
  cond[2] = (float)(l & 31) * (1.f / 31.f);
  cond[3] = 1.f - cond[2];
  #pragma unroll
  for (int t = 0; t < CLEN_; ++t) cond[4 + t] = cpar[t];

  float sD[9];
  float M = -3.0e38f;
  #pragma unroll
  for (int k = 0; k < 9; ++k) {
    float q = 0.f;
    #pragma unroll
    for (int j = 0; j < NCOND; ++j) q += cond[j] * WqD[k * NCOND + j];
    float s = q * kD[k];
    sD[k] = s;
    M = fmaxf(M, s);
  }
  float Z = 0.f, e[9];
  #pragma unroll
  for (int k = 0; k < 9; ++k) { e[k] = __expf(sD[k] - M); Z += e[k]; }
  float invZ = a[0] / Z;
  float adj[9];
  #pragma unroll
  for (int k = 0; k < 9; ++k) adj[k] = kernelD[k] + invZ * e[k] * vD[k];

  int y0 = l >> 5, x0 = l & 31;
  for (int c = 0; c < NC; ++c) {
    const float* xb = x + (((size_t)b * NC + c) << 10);
    float s = 0.f;
    #pragma unroll
    for (int ki = 0; ki < 3; ++ki) {
      int yy = y0 + ki - 1;
      if (yy < 0 || yy > 31) continue;
      #pragma unroll
      for (int kj = 0; kj < 3; ++kj) {
        int xx = x0 + kj - 1;
        if (xx < 0 || xx > 31) continue;
        s += xb[(yy << 5) + xx] * adj[ki * 3 + kj];
      }
    }
    xs[(((size_t)b * NC + c) << 10) + l] = s;
  }
}

// ---------- K5: fused pointwise attention + output ----------
// One block = 16 locations. Phase 1: software-pipelined WMMA logits (f16) into LDS.
// Phase 2: per-location softmax stats (vectorized b128 LDS reads).
// Phase 3: fused contraction into an LDS output tile; coalesced final store.
__global__ __launch_bounds__(256) void k_main(
    const float* __restrict__ kernelP,
    const float* __restrict__ vP,
    const float* __restrict__ xs,
    const _Float16* __restrict__ Apack,
    const float* __restrict__ a,
    const float* __restrict__ cpar,
    const float* __restrict__ bias,
    float* __restrict__ out) {
  __shared__ __align__(16) _Float16 sbuf[16 * SSTR];  // 147,712 B logits
  __shared__ float xsl[16 * NC];                      //   3,072 B
  __shared__ float ytile[16 * YSTR];                  //   6,208 B

  int tid  = threadIdx.x;
  int lane = tid & 31;
  int wid  = tid >> 5;
  int g0   = blockIdx.x * 16;
  int b    = g0 >> 10;
  int l0   = g0 & 1023;

  // stage xs tile: xsl[n*NC + c]
  for (int i = tid; i < 16 * NC; i += 256) {
    int n = i / NC, c = i - n * NC;
    xsl[i] = xs[(((size_t)b * NC + c) << 10) + l0 + n];
  }

  // B operand (32x16 f16): lane n<16 holds cond[l0+n][K=0..15] (K>=12 zero);
  // lanes 16-31 hold K=16..31 -> all zero.
  v16h bv = {};
  if (lane < 16) {
    int l = l0 + lane;
    float cond[NCOND];
    cond[0] = (float)(l >> 5) * (1.f / 31.f);
    cond[1] = 1.f - cond[0];
    cond[2] = (float)(l & 31) * (1.f / 31.f);
    cond[3] = 1.f - cond[2];
    #pragma unroll
    for (int t = 0; t < CLEN_; ++t) cond[4 + t] = cpar[t];
    #pragma unroll
    for (int h = 0; h < NCOND; ++h) bv[h] = (_Float16)cond[h];
  }

  // Phase 1: 36 tiles per wave, software-pipelined A loads (hide L2 latency at
  // 2-waves/SIMD occupancy). All control is wave-uniform: EXEC stays full for WMMA.
  v8f cz = {};
  int n1 = lane & 15;
  int mb = (lane >> 4) << 3;
  size_t aoff = ((size_t)wid * 32 + lane) << 4;      // element (half) offset
  const size_t astep = (size_t)(8 * 32) << 4;        // 8 tiles ahead
  v16h av = *(const v16h*)(Apack + aoff);
  for (int it = 0; it < NTILE / 8; ++it) {
    int T = wid + (it << 3);
    size_t anext = (it == NTILE / 8 - 1) ? aoff : (aoff + astep);  // last: dummy reload
    v16h avn = *(const v16h*)(Apack + anext);        // issue next load early
    v8f d = __builtin_amdgcn_wmma_f32_16x16x32_f16(false, av, false, bv,
                                                   (short)0, cz, false, false);
    v8h hv;
    #pragma unroll
    for (int r = 0; r < 8; ++r) hv[r] = (_Float16)d[r];
    // D layout: VGPR r -> row M = mb + r, col N = n1
    *(v8h*)(sbuf + n1 * SSTR + T * 16 + mb) = hv;    // 16B-aligned ds_store_b128
    av = avn;
    aoff = anext;
  }
  __syncthreads();

  // softmax stats: 16 threads per location; contiguous v8h chunks (ds_load_b128)
  int n2 = tid >> 4, t16 = tid & 15;
  const _Float16* sr = sbuf + n2 * SSTR;
  float M = -3.0e38f;
  for (int i0 = t16 * 8; i0 < CO; i0 += 128) {       // 36 chunks per thread
    v8h h = *(const v8h*)(sr + i0);
    #pragma unroll
    for (int r = 0; r < 8; ++r) M = fmaxf(M, (float)h[r]);
  }
  #pragma unroll
  for (int m = 8; m >= 1; m >>= 1) M = fmaxf(M, __shfl_xor(M, m, 32));
  float Z = 0.f;
  for (int i0 = t16 * 8; i0 < CO; i0 += 128) {
    v8h h = *(const v8h*)(sr + i0);
    #pragma unroll
    for (int r = 0; r < 8; ++r) Z += __expf((float)h[r] - M);
  }
  #pragma unroll
  for (int m = 8; m >= 1; m >>= 1) Z += __shfl_xor(Z, m, 32);

  // fused output: 12 threads x 8 contiguous o per location
  float inv = a[1] / Z;
  const float* xrow = xsl + n2 * NC;
  int o0 = t16 * 8;
  if (t16 < 12) {
    float acc[8];
    #pragma unroll
    for (int r = 0; r < 8; ++r) acc[r] = 0.f;
    for (int c = 0; c < NC; ++c) {
      int co = c * NO + o0;
      v8h sh = *(const v8h*)(sr + co);                       // ds_load_b128
      const float4* kp4 = (const float4*)(kernelP + co);     // coalesced b128 x2
      const float4* vp4 = (const float4*)(vP + co);
      float4 k0 = kp4[0], k1 = kp4[1];
      float4 v0 = vp4[0], v1 = vp4[1];
      float kk[8] = {k0.x, k0.y, k0.z, k0.w, k1.x, k1.y, k1.z, k1.w};
      float vv[8] = {v0.x, v0.y, v0.z, v0.w, v1.x, v1.y, v1.z, v1.w};
      float xc = xrow[c];
      #pragma unroll
      for (int r = 0; r < 8; ++r)
        acc[r] += xc * (kk[r] + inv * __expf((float)sh[r] - M) * vv[r]);
    }
    #pragma unroll
    for (int r = 0; r < 8; ++r) ytile[n2 * YSTR + o0 + r] = acc[r];
  }
  __syncthreads();

  // coalesced store: 16 consecutive l per o
  for (int i = tid; i < NO * 16; i += 256) {
    int o = i >> 4, n = i & 15;
    out[(((size_t)b * NO + o) << 10) + l0 + n] = ytile[n * YSTR + o] + bias[o];
  }
}

extern "C" void kernel_launch(void* const* d_in, const int* in_sizes, int n_in,
                              void* d_out, int out_size, void* d_ws, size_t ws_size,
                              hipStream_t stream) {
  const float* x       = (const float*)d_in[0];
  const float* kernelD = (const float*)d_in[1];
  const float* kernelP = (const float*)d_in[2];
  const float* bias    = (const float*)d_in[3];
  const float* a       = (const float*)d_in[4];
  const float* cpar    = (const float*)d_in[5];
  const float* WkD     = (const float*)d_in[6];
  const float* WvD     = (const float*)d_in[7];
  const float* WqD     = (const float*)d_in[8];
  const float* WkP     = (const float*)d_in[9];
  const float* WvP     = (const float*)d_in[10];
  const float* WqP     = (const float*)d_in[11];
  float* out = (float*)d_out;

  // workspace layout (bytes):
  //   [0,64)        kD[9]
  //   [64,128)      vD[9]
  //   [128, +18432) kP[4608]
  //   [.., +18432)  vP[4608]
  //   [.., +1.5MB)  xs[8*48*1024]
  //   [.., +288KB)  Apack f16 [288 tiles][32 lanes][16]  (32B aligned)
  char* ws = (char*)d_ws;
  float*    kD    = (float*)(ws + 0);
  float*    vD    = (float*)(ws + 64);
  float*    kP    = (float*)(ws + 128);
  float*    vP    = (float*)(ws + 128 + CO * 4);
  float*    xsbuf = (float*)(ws + 128 + 2 * CO * 4);
  _Float16* Apack = (_Float16*)(ws + 128 + 2 * CO * 4 + (size_t)NB * NC * NL * 4);

  k_prep_small<<<1, 32, 0, stream>>>(WkD, WvD, kernelD, kD, vD);
  k_matvec_big<<<(2 * CO) / 8, 256, 0, stream>>>(WkP, WvP, kernelP, kP, vP);
  k_pack_A<<<NTILE, 32, 0, stream>>>(WqP, kP, Apack);
  k_depthwise<<<(NB * NL) / 256, 256, 0, stream>>>(x, kernelD, WqD, kD, vD, a, cpar, xsbuf);
  k_main<<<(NB * NL) / 16, 256, 0, stream>>>(kernelP, vP, xsbuf, Apack, a, cpar, bias, out);

  (void)in_sizes; (void)n_in; (void)out_size; (void)ws_size;
}
